// LSTM_69973607186942
// MI455X (gfx1250) — compile-verified, matrix operand order checked by reading, not run on hardware
//
#include <hip/hip_runtime.h>
#include <hip/hip_bf16.h>

// ---------------- problem constants ----------------
#define HSZ   200          // hidden size
#define TSEQ  8192         // sequence length
#define NWG   8            // persistent workgroups (each owns 25 hidden channels)
#define JS    25           // hidden channels per WG
#define ROWS  100          // 4 gates * JS output rows per WG per matrix
#define NT    7            // N tiles of 16 (112 >= 100)
#define KTA   7            // K tiles of 32 for layer-0 recurrent GEMV (224 >= 200)
#define KTB   13           // K tiles of 32 for layer-1 concat GEMV  (416 >= 400)
#define BLK   256          // threads per WG (8 waves of 32)

typedef __attribute__((ext_vector_type(16))) __bf16 v16bf;
typedef __attribute__((ext_vector_type(8)))  float  v8f;

union AFrag { v16bf v; unsigned u[8]; };

__device__ inline unsigned short f2bf(float f) {
  unsigned u = __float_as_uint(f);
  unsigned rnd = ((u >> 16) & 1u) + 0x7FFFu;   // round-to-nearest-even
  return (unsigned short)((u + rnd) >> 16);
}

__device__ inline float fsig(float x) { return 1.0f / (1.0f + __expf(-x)); }
__device__ inline float ftanh(float x) {
  float ax = fabsf(x);
  float e  = __expf(-2.0f * ax);
  float t  = (1.0f - e) / (1.0f + e);
  return copysignf(t, x);
}

// generation-counter grid barrier across NWG persistent workgroups
__device__ inline void gridSync(unsigned* bar) {
  __syncthreads();
  if (threadIdx.x == 0) {
    unsigned gen = __hip_atomic_load(&bar[1], __ATOMIC_RELAXED, __HIP_MEMORY_SCOPE_AGENT);
    unsigned arr = __hip_atomic_fetch_add(&bar[0], 1u, __ATOMIC_ACQ_REL, __HIP_MEMORY_SCOPE_AGENT);
    if (arr == NWG - 1) {
      __hip_atomic_store(&bar[0], 0u, __ATOMIC_RELAXED, __HIP_MEMORY_SCOPE_AGENT);
      __hip_atomic_fetch_add(&bar[1], 1u, __ATOMIC_RELEASE, __HIP_MEMORY_SCOPE_AGENT);
    } else {
      while (__hip_atomic_load(&bar[1], __ATOMIC_ACQUIRE, __HIP_MEMORY_SCOPE_AGENT) == gen)
        __builtin_amdgcn_s_sleep(1);
    }
  }
  __syncthreads();
}

__global__ void lstm_init(unsigned* bar, float* hg) {
  int i = threadIdx.x;
  if (i < 8) bar[i] = 0u;
  for (int k = i; k < 2 * HSZ; k += BLK) hg[k] = 0.0f;
}

__global__ __launch_bounds__(BLK, 1)
void lstm_persistent(const float* __restrict__ x,
                     const float* __restrict__ W_ih0,
                     const float* __restrict__ W_hh0,
                     const float* __restrict__ b_ih0,
                     const float* __restrict__ b_hh0,
                     const float* __restrict__ W_ih1,
                     const float* __restrict__ W_hh1,
                     const float* __restrict__ b_ih1,
                     const float* __restrict__ b_hh1,
                     const float* __restrict__ W_out,
                     const float* __restrict__ b_out,
                     float* __restrict__ out,
                     unsigned* __restrict__ bar,
                     float* __restrict__ hg0,
                     float* __restrict__ hg1) {
  // ---- LDS: WMMA B-fragments for the WG's weight slices (bf16, pre-swizzled) ----
  __shared__ __align__(32) unsigned sWA[NT * KTA * 256];   // layer0 W_hh slice  (~49 KB)
  __shared__ __align__(32) unsigned sWB[NT * KTB * 256];   // layer1 [W_ih;W_hh] (~91 KB)
  __shared__ __align__(32) unsigned sHcat[KTB * 16];       // bf16 [h0(200); h1(200); pad]
  __shared__ float sGpre[NT * 16];
  __shared__ float sWih0[ROWS], sB0[ROWS], sB1[ROWS];
  __shared__ float sC0[JS], sC1[JS];
  __shared__ float sRed[BLK];

  const int wg   = blockIdx.x;
  const int tid  = threadIdx.x;
  const int lane = tid & 31;
  const int wid  = tid >> 5;
  const int j0   = wg * JS;

  // ---------------- one-time staging ----------------
  for (int r = tid; r < ROWS; r += BLK) {
    int g = r / JS, jj = r % JS;
    int grow = g * HSZ + (j0 + jj);
    sWih0[r] = W_ih0[grow];                      // (4H,1)
    sB0[r]   = b_ih0[grow] + b_hh0[grow];
    sB1[r]   = b_ih1[grow] + b_hh1[grow];
  }
  for (int d = tid; d < KTB * 16; d += BLK) sHcat[d] = 0u;  // h0=h1=0 at t=0 (+zero pad)
  if (tid < JS) { sC0[tid] = 0.0f; sC1[tid] = 0.0f; }

  // Swizzle layer-0 recurrent weights into WMMA B fragments.
  // Fragment layout (per ISA 16-bit operand striping): lane = {n = lane&15, khalf = lane>>4},
  // dword r<4 holds K = khalf*8 + 2r (+1); dword r>=4 holds K = 16 + khalf*8 + 2(r-4) (+1).
  for (int d = tid; d < NT * KTA * 256; d += BLK) {
    int frag = d >> 8, rem = d & 255;
    int fl = rem >> 3, fr = rem & 7;
    int ntile = frag / KTA, ktile = frag % KTA;
    int n = fl & 15, khalf = fl >> 4;
    int klo = (fr < 4) ? (2 * fr) : (16 + 2 * (fr - 4));
    int k = ktile * 32 + khalf * 8 + klo;
    int row = ntile * 16 + n;
    float v0 = 0.0f, v1 = 0.0f;
    if (row < ROWS) {
      int g = row / JS, jj = row % JS;
      const float* wr = W_hh0 + (size_t)(g * HSZ + j0 + jj) * HSZ;
      if (k     < HSZ) v0 = wr[k];
      if (k + 1 < HSZ) v1 = wr[k + 1];
    }
    sWA[d] = (unsigned)f2bf(v0) | ((unsigned)f2bf(v1) << 16);
  }
  // Layer-1: K-concat of W_ih1 (k<200, input h0) and W_hh1 (k-200, input h1).
  for (int d = tid; d < NT * KTB * 256; d += BLK) {
    int frag = d >> 8, rem = d & 255;
    int fl = rem >> 3, fr = rem & 7;
    int ntile = frag / KTB, ktile = frag % KTB;
    int n = fl & 15, khalf = fl >> 4;
    int klo = (fr < 4) ? (2 * fr) : (16 + 2 * (fr - 4));
    int k = ktile * 32 + khalf * 8 + klo;
    int row = ntile * 16 + n;
    float v0 = 0.0f, v1 = 0.0f;
    if (row < ROWS) {
      int g = row / JS, jj = row % JS;
      size_t grow = (size_t)(g * HSZ + j0 + jj);
      const float* wi = W_ih1 + grow * HSZ;
      const float* wh = W_hh1 + grow * HSZ;
      int k1 = k + 1;
      if (k  < 2 * HSZ) v0 = (k  < HSZ) ? wi[k]  : wh[k  - HSZ];
      if (k1 < 2 * HSZ) v1 = (k1 < HSZ) ? wi[k1] : wh[k1 - HSZ];
    }
    sWB[d] = (unsigned)f2bf(v0) | ((unsigned)f2bf(v1) << 16);
  }
  __syncthreads();

  unsigned short* hc16 = reinterpret_cast<unsigned short*>(sHcat);
  const int n16   = lane & 15;
  const int khalf = lane >> 4;
  const int myrow = wid * 16 + n16;

  // ---------------- recurrent time loop ----------------
  for (int t = 0; t < TSEQ; ++t) {
    float x_t = x[t];
    if (t + 16 < TSEQ) __builtin_prefetch(&x[t + 16], 0, 0);

    // ---- stage A: gates0 = pre0 + W_hh0 * h0  (K = 200) ----
    if (wid < NT) {
      float bias = (myrow < ROWS) ? (x_t * sWih0[myrow] + sB0[myrow]) : 0.0f;
      v8f acc = {bias, bias, bias, bias, bias, bias, bias, bias};
      for (int kt = 0; kt < KTA; ++kt) {
        AFrag a;
        int b0d = kt * 16 + 4 * khalf;   // dword index of bf16 (kt*32 + 8*khalf)
        a.u[0] = sHcat[b0d + 0];  a.u[1] = sHcat[b0d + 1];
        a.u[2] = sHcat[b0d + 2];  a.u[3] = sHcat[b0d + 3];
        a.u[4] = sHcat[b0d + 8];  a.u[5] = sHcat[b0d + 9];
        a.u[6] = sHcat[b0d + 10]; a.u[7] = sHcat[b0d + 11];
        const v16bf bf = *reinterpret_cast<const v16bf*>(&sWA[(wid * KTA + kt) * 256 + lane * 8]);
        acc = __builtin_amdgcn_wmma_f32_16x16x32_bf16(false, a.v, false, bf,
                                                      (short)0, acc, false, false);
      }
      if (lane < 16) sGpre[myrow] = acc[0];
    }
    __syncthreads();
    if (tid < JS) {
      float ig = fsig (sGpre[tid]);
      float fg = fsig (sGpre[JS + tid]);
      float gg = ftanh(sGpre[2 * JS + tid]);
      float og = fsig (sGpre[3 * JS + tid]);
      float c  = fg * sC0[tid] + ig * gg;
      sC0[tid] = c;
      hg0[j0 + tid] = og * ftanh(c);
    }
    gridSync(bar);
    for (int k = tid; k < HSZ; k += BLK) hc16[k] = f2bf(hg0[k]);
    __syncthreads();

    // ---- stage B: gates1 = b1 + [W_ih1;W_hh1] * [h0;h1]  (K = 400) ----
    if (wid < NT) {
      float bias = (myrow < ROWS) ? sB1[myrow] : 0.0f;
      v8f acc = {bias, bias, bias, bias, bias, bias, bias, bias};
      for (int kt = 0; kt < KTB; ++kt) {
        AFrag a;
        int b0d = kt * 16 + 4 * khalf;
        a.u[0] = sHcat[b0d + 0];  a.u[1] = sHcat[b0d + 1];
        a.u[2] = sHcat[b0d + 2];  a.u[3] = sHcat[b0d + 3];
        a.u[4] = sHcat[b0d + 8];  a.u[5] = sHcat[b0d + 9];
        a.u[6] = sHcat[b0d + 10]; a.u[7] = sHcat[b0d + 11];
        const v16bf bf = *reinterpret_cast<const v16bf*>(&sWB[(wid * KTB + kt) * 256 + lane * 8]);
        acc = __builtin_amdgcn_wmma_f32_16x16x32_bf16(false, a.v, false, bf,
                                                      (short)0, acc, false, false);
      }
      if (lane < 16) sGpre[myrow] = acc[0];
    }
    __syncthreads();
    if (tid < JS) {
      float ig = fsig (sGpre[tid]);
      float fg = fsig (sGpre[JS + tid]);
      float gg = ftanh(sGpre[2 * JS + tid]);
      float og = fsig (sGpre[3 * JS + tid]);
      float c  = fg * sC1[tid] + ig * gg;
      sC1[tid] = c;
      hg1[j0 + tid] = og * ftanh(c);
    }
    gridSync(bar);
    for (int k = tid; k < HSZ; k += BLK) hc16[HSZ + k] = f2bf(hg1[k]);
    __syncthreads();
  }

  // ---------------- output head: out = W_out . h1 + b_out ----------------
  if (wg == 0) {
    float p = 0.0f;
    for (int k = tid; k < HSZ; k += BLK) p += hg1[k] * W_out[k];
    sRed[tid] = p;
    __syncthreads();
    for (int s = BLK / 2; s > 0; s >>= 1) {
      if (tid < s) sRed[tid] += sRed[tid + s];
      __syncthreads();
    }
    if (tid == 0) out[0] = sRed[0] + b_out[0];
  }
}

extern "C" void kernel_launch(void* const* d_in, const int* in_sizes, int n_in,
                              void* d_out, int out_size, void* d_ws, size_t ws_size,
                              hipStream_t stream) {
  (void)in_sizes; (void)n_in; (void)out_size; (void)ws_size;
  const float* x     = (const float*)d_in[0];
  const float* W_ih0 = (const float*)d_in[1];
  const float* W_hh0 = (const float*)d_in[2];
  const float* b_ih0 = (const float*)d_in[3];
  const float* b_hh0 = (const float*)d_in[4];
  const float* W_ih1 = (const float*)d_in[5];
  const float* W_hh1 = (const float*)d_in[6];
  const float* b_ih1 = (const float*)d_in[7];
  const float* b_hh1 = (const float*)d_in[8];
  const float* W_out = (const float*)d_in[9];
  const float* b_out = (const float*)d_in[10];

  unsigned* bar = (unsigned*)d_ws;             // [0]=count, [1]=generation
  float* hg0 = (float*)d_ws + 16;              // h0 broadcast (200 fp32)
  float* hg1 = hg0 + 256;                      // h1 broadcast (200 fp32)

  lstm_init<<<1, BLK, 0, stream>>>(bar, hg0);
  lstm_persistent<<<NWG, BLK, 0, stream>>>(x, W_ih0, W_hh0, b_ih0, b_hh0,
                                           W_ih1, W_hh1, b_ih1, b_hh1,
                                           W_out, b_out,
                                           (float*)d_out, bar, hg0, hg1);
}
